// DiscreteContinuousDecoder_48069273977526
// MI455X (gfx1250) — matrix operand order repaired and checked by reading, not compiled, not run on hardware
//
#include <hip/hip_runtime.h>
#include <hip/hip_bf16.h>
#include <math.h>

// ---------------- problem constants (match reference) ----------------
#define NLAT_I 240
#define NLON_I 480
#define NLAT_O 361
#define NLON_O 720
#define CI     32
#define CO     32
#define NR     3
#define NPHI   3
#define KH     7          // K = (NR-1)*NPHI + 1
#define WOFF   10         // ceil(CUTOFF / (pi/360)) = ceil(9.75)
#define WLAT   21         // 2*WOFF + 1
#define CKDIM  (CI * KH)  // 224 = contracted dim of final GEMM

#define PI_F     3.14159265358979323846f
#define TWO_PI_F 6.28318530717958647692f
#define CUTOFF_F ((NR + 0.25f) * PI_F / 120.0f)   // 0.0850848...
#define DR_F     (CUTOFF_F / NR)
#define DPH_F    (TWO_PI_F / NPHI)

#define PSI_KS   (NLAT_O * WLAT * NLON_O)         // 5,458,320 per-k stride
#define PSI_TOT  (KH * PSI_KS)                    // 38,208,240
#define XU_TOT   (CI * NLAT_O * NLON_O)           // 8,317,440

// GEMM tiling: one block = one latitude x 48 longitudes.
#define PB 48                 // 720 / 48 = 15 tiles, 3 N-tiles of 16
#define THREADS 192           // 6 wave32 = 2 M-tiles x 3 N-tiles

// gfx1250 async Global->LDS path (guarded: falls back to plain LDS stores)
#if defined(__has_builtin)
#if __has_builtin(__builtin_amdgcn_global_load_async_to_lds_b32)
#define HAVE_ASYNC_LDS 1
#endif
#endif

#if defined(HAVE_ASYNC_LDS)
typedef __attribute__((address_space(1))) int* gint_p;
typedef __attribute__((address_space(3))) int* lint_p;
#if __has_builtin(__builtin_amdgcn_s_wait_asynccnt)
#define WAIT_ASYNC() __builtin_amdgcn_s_wait_asynccnt(0)
#else
#define WAIT_ASYNC() asm volatile("s_wait_asynccnt 0x0" ::: "memory")
#endif
#endif

typedef __attribute__((ext_vector_type(16))) _Float16 v16h;
typedef __attribute__((ext_vector_type(8)))  float    v8f;

// ---------------- kernel 1: bilinear upsample ----------------
__global__ void k_upsample(const float* __restrict__ x, float* __restrict__ xu) {
  int idx = blockIdx.x * blockDim.x + threadIdx.x;
  if (idx >= XU_TOT) return;
  int p = idx % NLON_O;
  int t = (idx / NLON_O) % NLAT_O;
  int i = idx / (NLON_O * NLAT_O);

  float pos_t = (float)t * (float)(NLAT_I - 1) / (float)(NLAT_O - 1); // t*239/360
  int i0 = (int)floorf(pos_t);
  i0 = min(max(i0, 0), NLAT_I - 1);
  int i1 = min(i0 + 1, NLAT_I - 1);
  float wt = pos_t - (float)i0;

  float pos_p = (float)p * (float)NLON_I / (float)NLON_O;             // p*2/3
  float j0f = floorf(pos_p);
  int j0 = ((int)j0f) % NLON_I;
  int j1 = (j0 + 1) % NLON_I;
  float wp = pos_p - j0f;

  const float* xi = x + (size_t)i * NLAT_I * NLON_I;
  float v00 = xi[i0 * NLON_I + j0];
  float v01 = xi[i0 * NLON_I + j1];
  float v10 = xi[i1 * NLON_I + j0];
  float v11 = xi[i1 * NLON_I + j1];
  xu[idx] = (1.f - wt) * ((1.f - wp) * v00 + wp * v01)
          +         wt * ((1.f - wp) * v10 + wp * v11);
}

// ---------------- kernel 2a: init nonzero-band widths ----------------
__global__ void k_init_maxw(int* __restrict__ maxw) {
  int idx = blockIdx.x * blockDim.x + threadIdx.x;
  if (idx < NLAT_O * WLAT) maxw[idx] = -1;
}

// ---------------- kernel 2b: build psi (unnormalized, q*mask applied) ----------------
__global__ void k_build_psi(float* __restrict__ psi, int* __restrict__ maxw) {
  int idx = blockIdx.x * blockDim.x + threadIdx.x;
  if (idx >= PSI_KS) return;
  int p = idx % NLON_O;
  int d = (idx / NLON_O) % WLAT;
  int t = idx / (NLON_O * WLAT);

  int lat_in = t - WOFF + d;
  bool valid = (lat_in >= 0) && (lat_in < NLAT_O);
  int latc = min(max(lat_in, 0), NLAT_O - 1);

  float theta_t = (float)t * (PI_F / (float)(NLAT_O - 1));
  float gamma   = (float)latc * (PI_F / (float)(NLAT_O - 1));
  float alpha   = -theta_t;
  float beta    = (float)p * (TWO_PI_F / (float)NLON_O);

  float ca = cosf(alpha), sa = sinf(alpha);
  float cg = cosf(gamma), sg = sinf(gamma);
  float cb = cosf(beta),  sb = sinf(beta);

  float zz = ca * cg - cb * sa * sg;
  float xx = ca * cb * sg + sa * cg;
  float yy = sb * sg;
  float r  = acosf(fminf(fmaxf(zz, -1.f), 1.f));
  float phi = atan2f(yy, xx);
  phi = fmodf(phi, TWO_PI_F);
  if (phi < 0.f) phi += TWO_PI_F;

  float q = sinf(gamma) * (PI_F / (float)(NLAT_O - 1)) * (TWO_PI_F / (float)NLON_O);
  float mask = (valid && (r <= CUTOFF_F)) ? q : 0.f;

  // k = 0: radial hat at origin
  psi[idx] = fmaxf(0.f, 1.f - r / DR_F) * mask;
  #pragma unroll
  for (int k = 1; k < KH; ++k) {
    int ir = (k - 1) / NPHI + 1;
    int ip = (k - 1) % NPHI;
    float rad = fmaxf(0.f, 1.f - fabsf(r - (float)ir * DR_F) / DR_F);
    float dd  = fabsf(phi - (float)ip * DPH_F);
    dd = fminf(dd, TWO_PI_F - dd);
    float az  = fmaxf(0.f, 1.f - dd / DPH_F);
    psi[k * PSI_KS + idx] = rad * az * mask;
  }
  if (mask > 0.f) {
    int w = min(p, NLON_O - p);       // wrapped longitudinal distance from 0
    atomicMax(&maxw[t * WLAT + d], w);
  }
}

// ---------------- kernel 3: per-(k,t) reciprocal normalization scale ----------------
__global__ void k_scale(const float* __restrict__ psi, float* __restrict__ rscale) {
  __shared__ float red[256];
  int bid = blockIdx.x;                 // k*NLAT_O + t
  int k = bid / NLAT_O;
  int t = bid % NLAT_O;
  const float* base = psi + (size_t)k * PSI_KS + (size_t)t * (WLAT * NLON_O);
  float s = 0.f;
  for (int j = threadIdx.x; j < WLAT * NLON_O; j += 256) s += base[j];
  red[threadIdx.x] = s;
  __syncthreads();
  for (int off = 128; off > 0; off >>= 1) {
    if (threadIdx.x < off) red[threadIdx.x] += red[threadIdx.x + off];
    __syncthreads();
  }
  if (threadIdx.x == 0) rscale[bid] = 1.f / fmaxf(red[0], 1e-8f);
}

// ---------------- kernel 4: normalize psi in place ----------------
__global__ void k_normalize(float* __restrict__ psi, const float* __restrict__ rscale) {
  int idx = blockIdx.x * blockDim.x + threadIdx.x;
  if (idx >= PSI_TOT) return;
  int k = idx / PSI_KS;
  int t = (idx % PSI_KS) / (WLAT * NLON_O);
  psi[idx] *= rscale[k * NLAT_O + t];
}

// ---------------- kernel 5: fused sparse correlation + WMMA channel GEMM ----------------
// block = (t, 48-wide longitude tile).
// Stage A: for each latitude-window row d, async-stage the psi nonzero band
//          into LDS once per block, then accumulate 56 FMAs per band element
//          (8 channels x 7 bases) into registers.
// Stage B: y(32x48) = W(32x224) * z(224x48) via v_wmma_f32_16x16x32_f16.
__global__ __launch_bounds__(THREADS)
void k_disco(const float* __restrict__ xu, const float* __restrict__ psi,
             const int* __restrict__ maxw, const float* __restrict__ wg,
             float* __restrict__ y) {
  __shared__ _Float16 zT[PB][CKDIM + 8];      // [p'][ck], 22.3 KB
  __shared__ _Float16 Wl[CO][CKDIM + 8];      // [o][ck],  14.8 KB
  __shared__ float    band[NLON_O * 8];       // psi band [dp][k(pad 8)], 23 KB

  const int t   = blockIdx.x;
  const int pb  = blockIdx.y * PB;
  const int tid = threadIdx.x;

  // Preload weight into LDS as f16 (layout (CO,CI,K) flat == [o][ck], ck = i*7+k)
  for (int u = tid; u < CO * CKDIM; u += THREADS)
    Wl[u / CKDIM][u % CKDIM] = (_Float16)wg[u];

  // ---- Stage A ----
  const int pp0   = tid % PB;       // longitude within tile
  const int ibase = tid / PB;       // 0..3; thread owns channels ibase+4*ii
  float acc[8 * KH];
  #pragma unroll
  for (int u = 0; u < 8 * KH; ++u) acc[u] = 0.f;

  for (int d = 0; d < WLAT; ++d) {
    const int mw = maxw[t * WLAT + d];          // block-uniform
    if (mw < 0) continue;                       // no support on this row
    const int hi = (mw == NLON_O / 2) ? (NLON_O / 2 - 1) : mw;
    const int nb = hi + mw + 1;                 // band length (<= 720)
    const int row = min(max(t - WOFF + d, 0), NLAT_O - 1);
    const float* __restrict__ pr = psi + ((size_t)t * WLAT + d) * NLON_O;

    // stage psi band -> LDS (async Global->LDS on gfx1250)
    for (int e = tid; e < nb * KH; e += THREADS) {
      const int k  = e / nb;
      const int j  = e - k * nb;                // band index 0..nb-1
      const int dp = j - mw;
      const int pp = dp + ((dp < 0) ? NLON_O : 0);
      const float* gp = pr + (size_t)k * PSI_KS + pp;
#if defined(HAVE_ASYNC_LDS)
      __builtin_amdgcn_global_load_async_to_lds_b32(
          (gint_p)gp, (lint_p)&band[j * 8 + k], 0, 0);
#else
      band[j * 8 + k] = *gp;
#endif
    }
#if defined(HAVE_ASYNC_LDS)
    WAIT_ASYNC();
#endif
    __syncthreads();

    // 8 channel-row base pointers for this d
    const float* xr[8];
    #pragma unroll
    for (int ii = 0; ii < 8; ++ii)
      xr[ii] = xu + ((size_t)(ibase + 4 * ii) * NLAT_O + row) * NLON_O;
    __builtin_prefetch(xr[0] + pb, 0, 0);       // global_prefetch_b8

    int pq = pb + pp0 - mw;
    pq += (pq < 0) ? NLON_O : 0;                // in [0, 720)
    for (int j = 0; j < nb; ++j) {
      const float4 b0 = *(const float4*)&band[j * 8];
      const float4 b1 = *(const float4*)&band[j * 8 + 4];
      const float bv[KH] = {b0.x, b0.y, b0.z, b0.w, b1.x, b1.y, b1.z};
      float xv[8];
      #pragma unroll
      for (int ii = 0; ii < 8; ++ii) xv[ii] = xr[ii][pq];
      #pragma unroll
      for (int ii = 0; ii < 8; ++ii) {
        #pragma unroll
        for (int k = 0; k < KH; ++k)
          acc[ii * KH + k] = fmaf(bv[k], xv[ii], acc[ii * KH + k]);
      }
      ++pq;
      if (pq == NLON_O) pq = 0;
    }
    __syncthreads();                            // band reuse fence
  }

  // commit z tile to LDS as f16
  #pragma unroll
  for (int ii = 0; ii < 8; ++ii) {
    #pragma unroll
    for (int k = 0; k < KH; ++k)
      zT[pp0][(ibase + 4 * ii) * KH + k] = (_Float16)acc[ii * KH + k];
  }
  __syncthreads();

  // ---- Stage B: WMMA GEMM, 2x3 grid of 16x16 tiles, Kdim = 224 = 7 x 32 ----
  const int w    = tid >> 5;           // wave id 0..5
  const int lane = tid & 31;
  const int Mt   = w / 3;              // 0..1
  const int Nt   = w % 3;              // 0..2
  const int lm   = lane & 15;
  const int lh   = lane >> 4;

  v8f c = {};
  #pragma unroll
  for (int s = 0; s < CKDIM / 32; ++s) {
    const int kb = 32 * s;
    v16h a, b;
    #pragma unroll
    for (int j = 0; j < 16; ++j) {
      // A (16x32 f16): lane = row M; half j -> K = (j%8) + 16*(j/8) + 8*(lane/16)
      const int ka = kb + (j & 7) + 16 * (j >> 3) + 8 * lh;
      a[j] = Wl[Mt * 16 + lm][ka];
      // B (32x16 f16): lane = col N; half j -> K = (j%8) + 16*(j/8) + 8*(lane/16)
      const int kbb = kb + (j & 7) + 16 * (j >> 3) + 8 * lh;
      b[j] = zT[Nt * 16 + lm][kbb];
    }
    c = __builtin_amdgcn_wmma_f32_16x16x32_f16(false, a, false, b,
                                               (short)0, c, false, false);
  }

  // D layout: VGPR j -> M = j + 8*(lane/16), N = lane%16
  #pragma unroll
  for (int j = 0; j < 8; ++j) {
    const int o  = Mt * 16 + j + 8 * lh;
    const int pc = pb + Nt * 16 + lm;
    y[(size_t)o * NLAT_O * NLON_O + (size_t)t * NLON_O + pc] = c[j];
  }
}

// ---------------- host launcher ----------------
extern "C" void kernel_launch(void* const* d_in, const int* in_sizes, int n_in,
                              void* d_out, int out_size, void* d_ws, size_t ws_size,
                              hipStream_t stream) {
  (void)in_sizes; (void)n_in; (void)out_size; (void)ws_size;
  const float* x  = (const float*)d_in[0];   // (1, 32, 240, 480) f32
  const float* wg = (const float*)d_in[1];   // (32, 32, 7) f32
  float* y = (float*)d_out;                  // (1, 32, 361, 720) f32

  // workspace layout (floats), 256-elt aligned blocks:
  //   xu: 8,317,440 | psi: 38,208,240 | rscale: 2,560 | maxw(int): 7,680
  float* xu     = (float*)d_ws;
  float* psi    = xu + XU_TOT;                       // XU_TOT % 256 == 0
  float* rscale = psi + ((PSI_TOT + 255) & ~255);
  int*   maxw   = (int*)(rscale + 2560);

  k_upsample <<<(XU_TOT + 255) / 256, 256, 0, stream>>>(x, xu);
  k_init_maxw<<<(NLAT_O * WLAT + 255) / 256, 256, 0, stream>>>(maxw);
  k_build_psi<<<(PSI_KS + 255) / 256, 256, 0, stream>>>(psi, maxw);
  k_scale    <<<KH * NLAT_O, 256, 0, stream>>>(psi, rscale);
  k_normalize<<<(PSI_TOT + 255) / 256, 256, 0, stream>>>(psi, rscale);

  dim3 grid(NLAT_O, NLON_O / PB);   // (361, 15)
  k_disco<<<grid, THREADS, 0, stream>>>(xu, psi, maxw, wg, y);
}
